// BayesianNetwork_455266533964
// MI455X (gfx1250) — matrix operand order repaired and checked by reading, not compiled, not run on hardware
//
#include <hip/hip_runtime.h>
#include <hip/hip_bf16.h>
#include <math.h>

// ---------------------------------------------------------------------------
// Types for CDNA5 WMMA (wave32): v_wmma_f32_16x16x32_bf16
// ---------------------------------------------------------------------------
typedef __bf16 v16bf __attribute__((ext_vector_type(16)));
typedef float  v8f   __attribute__((ext_vector_type(8)));

#define NHLOG2PI_F (-0.9189385332046727f)   // -0.5*log(2*pi)
#define PI_F       (3.14159265358979323846f)
#define REG_F      (1e-6f)

// ---------------------------------------------------------------------------
// CDNA5 async global->LDS staging (CDNA5 ISA):
// per-lane 16B copy straight into LDS, tracked by ASYNCcnt.
// ---------------------------------------------------------------------------
__device__ __forceinline__ void bnn_async_ld_b128(unsigned int lds_off, const void* gaddr) {
    asm volatile("global_load_async_to_lds_b128 %0, %1, off"
                 :: "v"(lds_off), "v"(gaddr) : "memory");
}
__device__ __forceinline__ void bnn_wait_async0() {
    asm volatile("s_wait_asynccnt 0" ::: "memory");
}

// ---------------------------------------------------------------------------
// Kernel: zero the scalar accumulator slots in workspace
// ---------------------------------------------------------------------------
__global__ void bnn_zero_sums(float* sums) {
    if (threadIdx.x < 16) sums[threadIdx.x] = 0.0f;
}

// ---------------------------------------------------------------------------
// Kernel: f32 -> bf16 bulk convert (activations only; weights convert in-flight)
// ---------------------------------------------------------------------------
__global__ __launch_bounds__(256)
void bnn_f32_to_bf16(const float* __restrict__ src, __bf16* __restrict__ dst, int n) {
    int i = (blockIdx.x * blockDim.x + threadIdx.x) * 4;
    if (i < n) {
        float4 v = *(const float4*)(src + i);
        dst[i + 0] = (__bf16)v.x;
        dst[i + 1] = (__bf16)v.y;
        dst[i + 2] = (__bf16)v.z;
        dst[i + 3] = (__bf16)v.w;
    }
}

// ---------------------------------------------------------------------------
// Kernel: grid-stride reduction with atomicAdd finish.
//   mode 0: sum x^2 ; mode 1: sum NHLOG2PI - log(softplus) ; mode 2: NHLOG2PI-0.5x^2
// ---------------------------------------------------------------------------
__global__ __launch_bounds__(256)
void bnn_reduce(const float* __restrict__ x, int n, float* __restrict__ dst, int mode) {
    __shared__ float sm[256];
    float s = 0.0f;
    for (int i = blockIdx.x * blockDim.x + threadIdx.x; i < n;
         i += gridDim.x * blockDim.x) {
        float v = x[i];
        if (mode == 0)      s += v * v;
        else if (mode == 1) s += NHLOG2PI_F - logf(log1pf(expf(v)));
        else                s += NHLOG2PI_F - 0.5f * v * v;
    }
    sm[threadIdx.x] = s;
    __syncthreads();
    for (int st = 128; st > 0; st >>= 1) {
        if ((int)threadIdx.x < st) sm[threadIdx.x] += sm[threadIdx.x + st];
        __syncthreads();
    }
    if (threadIdx.x == 0) atomicAdd(dst, sm[0]);
}

// ---------------------------------------------------------------------------
// WMMA GEMM:  out_bf16[M,N] = act( (Abf[M,K] @ Wf32[K,N]) * inv||w|| + bias[N]*inv||b|| )
//  - A is bf16 in HBM, staged to LDS with GLOBAL_LOAD_ASYNC_TO_LDS_B128 (ASYNCcnt)
//  - W is f32 in HBM (read once), converted to bf16 in flight, stored transposed in LDS
//  - Block = 256 threads = 8 waves; block tile 128x128, BK=32 (one wmma K step)
//  - Wave w: 32(M) x 64(N) strip -> 2x4 tiles of 16x16 -> 8 v8f accumulators
// ---------------------------------------------------------------------------
#define BM 128
#define BN 128
#define BK 32

__device__ __forceinline__ v8f wmma_bf16(v16bf a, v16bf b, v8f c) {
    return __builtin_amdgcn_wmma_f32_16x16x32_bf16(
        /*neg_a=*/false, a, /*neg_b=*/false, b,
        /*c_mod=*/(short)0, c, /*reuse_a=*/false, /*reuse_b=*/false);
}

__global__ __launch_bounds__(256)
void bnn_gemm_wmma(const __bf16* __restrict__ A, const float* __restrict__ W,
                   const float* __restrict__ bias,
                   const float* __restrict__ sums, int wslot, int bslot,
                   __bf16* __restrict__ out, int M, int N, int K, int relu) {
    __shared__ __bf16 Xs[BM][BK];   // 8 KB : A tile, row-major [m][k]
    __shared__ __bf16 Ws[BN][BK];   // 8 KB : W tile, TRANSPOSED [n][k]

    const int t    = threadIdx.x;
    const int wave = t >> 5;
    const int lane = t & 31;
    const int bm = blockIdx.y * BM;
    const int bn = blockIdx.x * BN;
    const int waveM = wave >> 1;     // 0..3 -> 32-row strip
    const int waveN = wave & 1;      // 0..1 -> 64-col strip

    v8f acc[2][4] = {};

    // ISA 16-bit fragment lane layout (CDNA5 ISA §7.12.2)
    const int mlo = lane & 15;
    const int kbA = (lane < 16) ? 0 : 8;
    const int kbB = (lane < 16) ? 0 : 16;

    // async A staging geometry: 16B per lane, 8 rows per instruction, 2 per wave
    const unsigned int xs_base = (unsigned int)(uintptr_t)&Xs[0][0];
    const int ar0 = wave * 16 + (lane >> 2);     // rows ar0 and ar0+8
    const int ach = (lane & 3) * 16;             // byte chunk within 64B row

    for (int k0 = 0; k0 < K; k0 += BK) {
        // ---- A tile: async global(bf16) -> LDS, no VGPR round-trip
        {
            const char* g0 = (const char*)(A + (size_t)(bm + ar0) * K + k0) + ach;
            const char* g1 = (const char*)(A + (size_t)(bm + ar0 + 8) * K + k0) + ach;
            bnn_async_ld_b128(xs_base + (unsigned)(ar0 * (BK * 2) + ach), g0);
            bnn_async_ld_b128(xs_base + (unsigned)((ar0 + 8) * (BK * 2) + ach), g1);
        }
        // ---- W tile: f32 stream, convert to bf16, store transposed [n][k]
        {
            const int k  = t >> 3;            // 0..31
            const int n0 = (t & 7) * 16;      // 0..112
            const float* src = W + (size_t)(k0 + k) * N + bn + n0;
            __builtin_prefetch(src + (size_t)BK * N, 0, 1);
            #pragma unroll
            for (int i = 0; i < 16; i += 4) {
                float4 v = *(const float4*)(src + i);
                Ws[n0 + i + 0][k] = (__bf16)v.x;
                Ws[n0 + i + 1][k] = (__bf16)v.y;
                Ws[n0 + i + 2][k] = (__bf16)v.z;
                Ws[n0 + i + 3][k] = (__bf16)v.w;
            }
        }
        bnn_wait_async0();
        __syncthreads();

        // ---- build fragments per ISA lane layout
        v16bf a[2], b[4];
        #pragma unroll
        for (int p = 0; p < 8; ++p) {
            const int kk = kbA + 2 * p + ((p >= 4) ? 8 : 0);
            #pragma unroll
            for (int i = 0; i < 2; ++i) {
                a[i][2 * p]     = Xs[waveM * 32 + i * 16 + mlo][kk];
                a[i][2 * p + 1] = Xs[waveM * 32 + i * 16 + mlo][kk + 1];
            }
            const int k2 = kbA + 2 * p + ((p >= 4) ? 8 : 0);
            #pragma unroll
            for (int j = 0; j < 4; ++j) {
                b[j][2 * p]     = Ws[waveN * 64 + j * 16 + mlo][k2];
                b[j][2 * p + 1] = Ws[waveN * 64 + j * 16 + mlo][k2 + 1];
            }
        }

        #pragma unroll
        for (int i = 0; i < 2; ++i)
            #pragma unroll
            for (int j = 0; j < 4; ++j)
                acc[i][j] = wmma_bf16(a[i], b[j], acc[i][j]);

        __syncthreads();
    }

    // ---- epilogue: fold vMF normalization + bias + relu; bf16 out
    const float invw = rsqrtf(sums[wslot]);
    const float invb = rsqrtf(sums[bslot]);
    const int mshift = (lane >= 16) ? 8 : 0;
    #pragma unroll
    for (int i = 0; i < 2; ++i) {
        #pragma unroll
        for (int j = 0; j < 4; ++j) {
            const int mbase = bm + waveM * 32 + i * 16 + mshift;
            const int nn    = bn + waveN * 64 + j * 16 + mlo;
            const float bterm = bias[nn] * invb;
            #pragma unroll
            for (int r = 0; r < 8; ++r) {
                float v = acc[i][j][r] * invw + bterm;
                if (relu) v = fmaxf(v, 0.0f);
                out[(size_t)(mbase + r) * N + nn] = (__bf16)v;
            }
        }
    }
}

// ---------------------------------------------------------------------------
// Layer 3 (5 outputs) + log_softmax. One wave (32 lanes) per batch row.
// ---------------------------------------------------------------------------
__global__ __launch_bounds__(256)
void bnn_layer3_logsoftmax(const __bf16* __restrict__ h2,
                           const float* __restrict__ w3,   // [5, N]
                           const float* __restrict__ b3,   // [5]
                           float* __restrict__ out,        // [rows, 5]
                           int N) {
    const int wave = threadIdx.x >> 5;
    const int lane = threadIdx.x & 31;
    const int row  = blockIdx.x * 8 + wave;
    const __bf16* h = h2 + (size_t)row * N;

    float acc[5] = {0.f, 0.f, 0.f, 0.f, 0.f};
    for (int i = lane; i < N; i += 32) {
        const float x = (float)h[i];
        #pragma unroll
        for (int o = 0; o < 5; ++o) acc[o] += x * w3[(size_t)o * N + i];
    }
    #pragma unroll
    for (int o = 0; o < 5; ++o) {
        #pragma unroll
        for (int off = 16; off > 0; off >>= 1)
            acc[o] += __shfl_xor(acc[o], off, 32);
    }
    if (lane == 0) {
        float logits[5], mx = -INFINITY;
        #pragma unroll
        for (int o = 0; o < 5; ++o) { logits[o] = acc[o] + b3[o]; mx = fmaxf(mx, logits[o]); }
        float se = 0.f;
        #pragma unroll
        for (int o = 0; o < 5; ++o) se += expf(logits[o] - mx);
        const float lse = mx + logf(se);
        #pragma unroll
        for (int o = 0; o < 5; ++o) out[(size_t)row * 5 + o] = logits[o] - lse;
    }
}

// ---------------------------------------------------------------------------
// Finalize scalars: log_var_post (d_out[5120]) and log_prior (d_out[5121])
// ---------------------------------------------------------------------------
__device__ __forceinline__ float bnn_log_C_vmf(float d, float k) {
    const float s   = 0.5f * d - 1.0f;
    const float x   = k / s;
    const float sq  = sqrtf(1.0f + x * x);
    const float eta = sq + logf(x) - log1pf(sq);
    const float lbi = s * eta - 0.5f * logf(2.0f * PI_F * s) - 0.5f * logf(sq);
    return d * NHLOG2PI_F + s * logf(k) - lbi;
}
__device__ __forceinline__ float bnn_log_surface_area(float dim) {
    const float h = (dim + 1.0f) * 0.5f;
    return logf(2.0f) + h * logf(PI_F) - lgammaf(h);
}

__global__ void bnn_finalize(const float* __restrict__ sums,
                             const float* lkw1, const float* lkb1,
                             const float* lkw2, const float* lkb2,
                             float* __restrict__ out2) {
    if (threadIdx.x != 0 || blockIdx.x != 0) return;
    const float dw = 4096.0f * 4096.0f;
    const float db = 4096.0f;
    const float kw1 = expf(*lkw1) + REG_F, kb1 = expf(*lkb1) + REG_F;
    const float kw2 = expf(*lkw2) + REG_F, kb2 = expf(*lkb2) + REG_F;
    // <mu,mu> == 1 after normalization
    const float lvp1 = kw1 + bnn_log_C_vmf(dw, kw1) + kb1 + bnn_log_C_vmf(db, kb1);
    const float lvp2 = kw2 + bnn_log_C_vmf(dw, kw2) + kb2 + bnn_log_C_vmf(db, kb2);
    const float lvp3 = sums[4] + sums[5];
    // reference bug preserved: bias prior also uses d_w
    const float lsa  = bnn_log_surface_area(dw);
    const float lp12 = -2.0f * lsa + -2.0f * lsa;
    const float lp3  = sums[6] + sums[7];
    out2[0] = lvp1 + lvp2 + lvp3;
    out2[1] = lp12 + lp3;
}

// ---------------------------------------------------------------------------
// Host launcher
// ---------------------------------------------------------------------------
extern "C" void kernel_launch(void* const* d_in, const int* in_sizes, int n_in,
                              void* d_out, int out_size, void* d_ws, size_t ws_size,
                              hipStream_t stream) {
    const int B = 1024, IN = 4096, H = 4096;
    const float* x      = (const float*)d_in[0];
    const float* w1_mu  = (const float*)d_in[1];
    const float* lkw1   = (const float*)d_in[2];
    const float* b1_mu  = (const float*)d_in[3];
    const float* lkb1   = (const float*)d_in[4];
    const float* w2_mu  = (const float*)d_in[5];
    const float* lkw2   = (const float*)d_in[6];
    const float* b2_mu  = (const float*)d_in[7];
    const float* lkb2   = (const float*)d_in[8];
    const float* w3_mu  = (const float*)d_in[9];
    const float* w3_rho = (const float*)d_in[10];
    const float* b3_mu  = (const float*)d_in[11];
    const float* b3_rho = (const float*)d_in[12];
    float* out = (float*)d_out;

    float*  sums = (float*)d_ws;                        // 16 scalar slots (64 floats reserved)
    __bf16* xbf  = (__bf16*)((char*)d_ws + 256);        // [1024,4096] bf16 : 8 MB
    __bf16* h1   = xbf + (size_t)B * IN;                // [1024,4096] bf16 : 8 MB
    __bf16* h2   = h1 + (size_t)B * H;                  // [1024,4096] bf16 : 8 MB

    bnn_zero_sums<<<1, 32, 0, stream>>>(sums);

    bnn_f32_to_bf16<<<(B * IN) / (256 * 4), 256, 0, stream>>>(x, xbf, B * IN);

    bnn_reduce<<<1024, 256, 0, stream>>>(w1_mu, H * IN, sums + 0, 0);
    bnn_reduce<<<32,   256, 0, stream>>>(b1_mu, H,      sums + 1, 0);
    bnn_reduce<<<1024, 256, 0, stream>>>(w2_mu, H * H,  sums + 2, 0);
    bnn_reduce<<<32,   256, 0, stream>>>(b2_mu, H,      sums + 3, 0);
    bnn_reduce<<<32,   256, 0, stream>>>(w3_rho, 5 * H, sums + 4, 1);
    bnn_reduce<<<1,    256, 0, stream>>>(b3_rho, 5,     sums + 5, 1);
    bnn_reduce<<<32,   256, 0, stream>>>(w3_mu,  5 * H, sums + 6, 2);
    bnn_reduce<<<1,    256, 0, stream>>>(b3_mu,  5,     sums + 7, 2);

    dim3 g1(H / BN, B / BM);   // (32, 8)
    bnn_gemm_wmma<<<g1, 256, 0, stream>>>(xbf, w1_mu, b1_mu, sums, 0, 1,
                                          h1, B, H, IN, 1);
    bnn_gemm_wmma<<<g1, 256, 0, stream>>>(h1,  w2_mu, b2_mu, sums, 2, 3,
                                          h2, B, H, H, 1);

    bnn_layer3_logsoftmax<<<B / 8, 256, 0, stream>>>(h2, w3_mu, b3_mu, out, H);

    bnn_finalize<<<1, 1, 0, stream>>>(sums, lkw1, lkb1, lkw2, lkb2, out + B * 5);
}
